// LocalPoolPointnet_13778255086349
// MI455X (gfx1250) — compile-verified, matrix operand order checked by reading, not run on hardware
//

#include <hip/hip_runtime.h>
#include <hip/hip_bf16.h>
#include <cstdint>
#include <cstddef>

// ---------------- problem constants (from reference) ----------------
#define BATCH 4
#define TPTS  65536
#define RESO  128
#define NBINS (RESO*RESO)          // 16384
#define HDIM  128
#define HD2   256
#define NPTS  (BATCH*TPTS)         // 262144
#define WAVES 4
#define BLOCK (WAVES*32)           // 128 threads / workgroup (wave32)
#define MROWS 32                   // points per wave (two 16-row WMMA tiles)
#define PTS_PER_WG (WAVES*MROWS)   // 128
#define NWG2 (NPTS/PTS_PER_WG)     // 2048
// k_head tiling (16 pts / wave)
#define NWG_HEAD (NPTS/(WAVES*16)) // 4096

typedef __attribute__((ext_vector_type(16))) __bf16    bfx16;
typedef __attribute__((ext_vector_type(8)))  float     fx8;
typedef __attribute__((ext_vector_type(8)))  unsigned  ux8;
typedef int v4i __attribute__((__vector_size__(16)));  // b128 payload type for async builtins
typedef __attribute__((address_space(1))) v4i* gv4p;   // global (AS1)
typedef __attribute__((address_space(3))) v4i* lv4p;   // LDS (AS3)

// ---------------- LDS layout for the big kernels (bytes) ----------------
#define W0L_OFF 0                      // 64 KB : W0 fragments (256x128 bf16)
#define W1L_OFF 65536                  // 32 KB : W1 fragments (128x128 bf16)
#define WSL_OFF 98304                  // 64 KB : Ws fragments (256x128 bf16)
#define XN_OFF  163840                 // 64 KB : x activations, [WAVES][32][256] bf16
#define HB_OFF  229376                 // 32 KB : relu(fc0) h, [WAVES][32][128] bf16
#define OB_OFF  262144                 // 32 KB : output staging, [WAVES][32][128] bf16
#define MISC_OFF 294912                //  2 KB : per-wave point cache / idx cache
#define SH_BIG  (MISC_OFF + 2048)      // ~290 KB (<= 320 KB WGP LDS)

// ---------------- scalar helpers ----------------
static __device__ __forceinline__ unsigned short f2bfu(float f) {
  unsigned u = __builtin_bit_cast(unsigned, f);
  unsigned r = (u + 0x7FFFu + ((u >> 16) & 1u)) >> 16;   // RNE
  return (unsigned short)r;
}
static __device__ __forceinline__ float bfu2f(unsigned short s) {
  unsigned u = ((unsigned)s) << 16;
  return __builtin_bit_cast(float, u);
}
// monotonic float<->u32 for atomicMax-based scatter-max
static __device__ __forceinline__ unsigned encMax(float f) {
  unsigned u = __builtin_bit_cast(unsigned, f);
  return (u & 0x80000000u) ? ~u : (u | 0x80000000u);
}
static __device__ __forceinline__ float decMax(unsigned e) {
  unsigned u = (e & 0x80000000u) ? (e ^ 0x80000000u) : ~e;
  return __builtin_bit_cast(float, u);
}

// ---------------- gfx1250 async global<->LDS copy (ASYNCcnt path) ----------------
static __device__ __forceinline__ void async_ld16(void* lds, const void* g) {
#if __has_builtin(__builtin_amdgcn_global_load_async_to_lds_b128)
  __builtin_amdgcn_global_load_async_to_lds_b128(
      (gv4p)const_cast<void*>(g), (lv4p)lds, 0, 0);
#else
  *(uint4*)lds = *(const uint4*)g;     // fallback: sync copy
#endif
}
static __device__ __forceinline__ void async_st16(void* g, const void* lds) {
#if __has_builtin(__builtin_amdgcn_global_store_async_from_lds_b128)
  __builtin_amdgcn_global_store_async_from_lds_b128(
      (gv4p)g, (lv4p)const_cast<void*>(lds), 0, 0);
#else
  *(uint4*)g = *(const uint4*)lds;     // fallback: sync copy
#endif
}
static __device__ __forceinline__ void wait_async0() {
  asm volatile("s_wait_asynccnt 0x0" ::: "memory");
}
static __device__ __forceinline__ void wait_ds0() {
  asm volatile("s_wait_dscnt 0x0" ::: "memory");
}

// cooperative: stage nUsh bf16 elements global -> LDS (16B chunks, uniform EXEC)
static __device__ __forceinline__ void stage_async(unsigned short* lds,
                                                   const unsigned short* __restrict__ g,
                                                   int nUsh) {
  int chunks = nUsh >> 3;
  for (int c = (int)threadIdx.x; c < chunks; c += BLOCK)
    async_ld16(lds + (size_t)c * 8, g + (size_t)c * 8);
}

// ---------------- WMMA fragment helpers ----------------
// A-matrix 16x32 bf16 layout (ISA 7.12.2): lane<16 -> M=lane, K={0..7,16..23};
// lane>=16 -> M=lane-16, K={8..15,24..31}. Loaded as 2x16B from LDS.
static __device__ __forceinline__ ux8 loadA_pairs(const unsigned short* row, int kt, bool hi) {
  int k0 = kt * 32 + (hi ? 8 : 0);
  int k1 = kt * 32 + (hi ? 24 : 16);
  uint4 lo = *(const uint4*)(row + k0);
  uint4 h4 = *(const uint4*)(row + k1);
  ux8 u;
  u[0] = lo.x; u[1] = lo.y; u[2] = lo.z; u[3] = lo.w;
  u[4] = h4.x; u[5] = h4.y; u[6] = h4.z; u[7] = h4.w;
  return u;
}
// packed bf16 relu on pairs: zero any half with sign bit set
static __device__ __forceinline__ ux8 relu_pk(ux8 u) {
#pragma unroll
  for (int i = 0; i < 8; ++i) {
    unsigned t = u[i] & 0x80008000u;
    unsigned m = t | (t - (t >> 15));
    u[i] &= ~m;
  }
  return u;
}
static __device__ __forceinline__ bfx16 asA(ux8 u) { return __builtin_bit_cast(bfx16, u); }

// B fragments pre-packed fragment-major: one aligned 32B LDS read per lane.
static __device__ __forceinline__ bfx16 loadB(const unsigned short* wf,
                                              int kt, int nt, int KT, int lane) {
  const bfx16* p = (const bfx16*)(const void*)(wf + (size_t)((nt * KT + kt) * 32 + lane) * 16);
  return *p;
}
static __device__ __forceinline__ fx8 wmma_bf16(bfx16 a, bfx16 b, fx8 c) {
  return __builtin_amdgcn_wmma_f32_16x16x32_bf16(false, a, false, b, (short)0, c, false, false);
}

// ---------------- fused ResnetBlockFC, one wave = 32 points ----------------
// xn: LDS bf16 [32][256] raw x (relu applied on the fly), hl/ob: LDS bf16 [32][128]
static __device__ __forceinline__ void wave_resblock32(
    const unsigned short* __restrict__ xn,
    unsigned short* __restrict__ hl,
    unsigned short* __restrict__ ob,
    const unsigned short* __restrict__ w0L,   // LDS-resident fragments
    const unsigned short* __restrict__ w1L,
    const unsigned short* __restrict__ wsL,
    const float* __restrict__ b0,
    const float* __restrict__ b1,
    unsigned short* __restrict__ netOut)      // global, 32 pts * 128 contiguous
{
  const int lane = (int)(threadIdx.x & 31u);
  const int m16  = lane & 15;
  const bool hi  = lane >= 16;
  const unsigned short* x0 = xn + m16 * HD2;
  const unsigned short* x1 = xn + (16 + m16) * HD2;

  // GEMM1: net = relu(x) @ W0 + b0 ; hl = relu(net). Each B frag feeds 2 WMMAs.
  for (int nt = 0; nt < 8; ++nt) {
    fx8 a0 = {}, a1 = {};
#pragma unroll
    for (int kt = 0; kt < 8; ++kt) {
      bfx16 b = loadB(w0L, kt, nt, 8, lane);
      a0 = wmma_bf16(asA(relu_pk(loadA_pairs(x0, kt, hi))), b, a0);
      a1 = wmma_bf16(asA(relu_pk(loadA_pairs(x1, kt, hi))), b, a1);
    }
    int n = nt * 16 + m16;
    float bias = b0[n];
#pragma unroll
    for (int r = 0; r < 8; ++r) {
      int m = (hi ? 8 : 0) + r;                // C/D layout: lane<16 -> M=r else M=8+r
      hl[m * HDIM + n]        = f2bfu(fmaxf(a0[r] + bias, 0.f));
      hl[(16 + m) * HDIM + n] = f2bfu(fmaxf(a1[r] + bias, 0.f));
    }
  }

  // GEMM2 (+shortcut fused): out = hl@W1 + b1 + x@Ws
  const unsigned short* h0 = hl + m16 * HDIM;
  const unsigned short* h1 = hl + (16 + m16) * HDIM;
  for (int nt = 0; nt < 8; ++nt) {
    fx8 a0 = {}, a1 = {};
#pragma unroll
    for (int kt = 0; kt < 4; ++kt) {
      bfx16 b = loadB(w1L, kt, nt, 4, lane);
      a0 = wmma_bf16(asA(loadA_pairs(h0, kt, hi)), b, a0);
      a1 = wmma_bf16(asA(loadA_pairs(h1, kt, hi)), b, a1);
    }
#pragma unroll
    for (int kt = 0; kt < 8; ++kt) {
      bfx16 b = loadB(wsL, kt, nt, 8, lane);
      a0 = wmma_bf16(asA(loadA_pairs(x0, kt, hi)), b, a0);
      a1 = wmma_bf16(asA(loadA_pairs(x1, kt, hi)), b, a1);
    }
    int n = nt * 16 + m16;
    float bias = b1[n];
#pragma unroll
    for (int r = 0; r < 8; ++r) {
      int m = (hi ? 8 : 0) + r;
      ob[m * HDIM + n]        = f2bfu(a0[r] + bias);
      ob[(16 + m) * HDIM + n] = f2bfu(a1[r] + bias);
    }
  }

  // flush staged output rows (contiguous 8KB) to global via async store engine
  wait_ds0();                                  // LDS writes visible to async engine
  for (int c = lane; c < MROWS * HDIM / 8; c += 32)
    async_st16(netOut + (size_t)c * 8, ob + (size_t)c * 8);
  wait_async0();
}

// ---------------- kernels ----------------

// repack f32 weight [nmat][K][128] -> bf16 WMMA B fragments, fragment-major
__global__ void k_prep_frag(const float* __restrict__ src, unsigned short* __restrict__ dst,
                            int K, int nmat) {
  int KT = K >> 5;
  long total = (long)nmat * K * HDIM;
  for (long i = blockIdx.x * (long)blockDim.x + threadIdx.x; i < total;
       i += (long)gridDim.x * blockDim.x) {
    int mat = (int)(i / ((long)K * HDIM));
    int rem = (int)(i - (long)mat * K * HDIM);
    int k = rem / HDIM, n = rem % HDIM;
    int nt = n >> 4, nn = n & 15;
    int kt = k >> 5, kk = k & 31;
    int lane, j;
    if      (kk < 8)  { lane = nn;      j = kk;      }
    else if (kk < 16) { lane = nn + 16; j = kk - 8;  }
    else if (kk < 24) { lane = nn;      j = kk - 8;  }
    else              { lane = nn + 16; j = kk - 16; }
    long d = (long)mat * K * HDIM + ((long)(nt * KT + kt) * 32 + lane) * 16 + j;
    dst[d] = f2bfu(src[i]);
  }
}

__global__ void k_fill_u32(unsigned* __restrict__ p, unsigned v, long n) {
  for (long i = blockIdx.x * (long)blockDim.x + threadIdx.x; i < n;
       i += (long)gridDim.x * blockDim.x)
    p[i] = v;
}

// stage 0: idx + fc_pos + resblock[0]
__global__ __launch_bounds__(BLOCK) void k_encode(
    const float* __restrict__ p,
    const float* __restrict__ fpw, const float* __restrict__ fpb,
    const unsigned short* __restrict__ w0f, const float* __restrict__ b0,
    const unsigned short* __restrict__ w1f, const float* __restrict__ b1,
    const unsigned short* __restrict__ wsf,
    int* __restrict__ gidx, unsigned short* __restrict__ net) {
  extern __shared__ char smem[];
  unsigned short* w0L = (unsigned short*)(smem + W0L_OFF);
  unsigned short* w1L = (unsigned short*)(smem + W1L_OFF);
  unsigned short* wsL = (unsigned short*)(smem + WSL_OFF);
  unsigned short* xnb = (unsigned short*)(smem + XN_OFF);
  unsigned short* hb  = (unsigned short*)(smem + HB_OFF);
  unsigned short* obb = (unsigned short*)(smem + OB_OFF);
  float* pbb          = (float*)(smem + MISC_OFF);
  const int w = (int)(threadIdx.x >> 5), lane = (int)(threadIdx.x & 31u);
  unsigned short* xn = xnb + w * MROWS * HD2;
  unsigned short* hl = hb  + w * MROWS * HDIM;
  unsigned short* ob = obb + w * MROWS * HDIM;
  float* pl = pbb + w * MROWS * 4;
  const int pt0 = blockIdx.x * PTS_PER_WG + w * MROWS;

  // kick off async staging of this stage's weights (shared by all 4 waves)
  stage_async(w0L, w0f, HD2 * HDIM);
  stage_async(w1L, w1f, HDIM * HDIM);
  stage_async(wsL, wsf, HD2 * HDIM);

  // meanwhile: per-point index + position cache (one point per lane)
  {
    const float* P = p + (size_t)(pt0 + lane) * 3;
    float px = P[0], py = P[1], pz = P[2];
    pl[lane * 4 + 0] = px; pl[lane * 4 + 1] = py; pl[lane * 4 + 2] = pz;
    float fx = fminf(fmaxf(px / 1.101f + 0.5f, 0.f), 0.999f);
    float fz = fminf(fmaxf(pz / 1.101f + 0.5f, 0.f), 0.999f);
    gidx[pt0 + lane] = (int)(fx * (float)RESO) + RESO * (int)(fz * (float)RESO);
  }
  // x = p @ fc_pos_w (3x256) + b, stored bf16 (relu applied later on the fly)
  for (int t = lane; t < MROWS * HD2; t += 32) {
    int m = t >> 8, n = t & 255;
    const float* pr = pl + m * 4;
    xn[t] = f2bfu(fpb[n] + pr[0] * fpw[n] + pr[1] * fpw[256 + n] + pr[2] * fpw[512 + n]);
  }
  wait_async0();
  __syncthreads();
  wave_resblock32(xn, hl, ob, w0L, w1L, wsL, b0, b1, net + (size_t)pt0 * HDIM);
}

// scatter-max of net into bins (monotonic-u32 atomicMax), 2 channels per thread
__global__ void k_scatter_max(const unsigned short* __restrict__ net,
                              const int* __restrict__ gidx,
                              unsigned* __restrict__ bins) {
  long total = (long)NPTS * (HDIM / 2);
  const unsigned* net32 = (const unsigned*)net;
  for (long i = blockIdx.x * (long)blockDim.x + threadIdx.x; i < total;
       i += (long)gridDim.x * blockDim.x) {
    int pt = (int)(i >> 6);
    int c2 = (int)(i & 63);
    int b  = pt >> 16;
    int bin = gidx[pt];
    unsigned v = net32[i];
    unsigned* dst = bins + ((size_t)b * NBINS + bin) * HDIM + c2 * 2;
    atomicMax(dst,     encMax(bfu2f((unsigned short)(v & 0xFFFFu))));
    atomicMax(dst + 1, encMax(bfu2f((unsigned short)(v >> 16))));
  }
}

// stages 1..4: gather pooled max, concat, resblock[i] (in-place on net)
__global__ __launch_bounds__(BLOCK) void k_block(
    const unsigned* __restrict__ bins,
    const int* __restrict__ gidx,
    const unsigned short* __restrict__ w0f, const float* __restrict__ b0,
    const unsigned short* __restrict__ w1f, const float* __restrict__ b1,
    const unsigned short* __restrict__ wsf,
    const unsigned short* __restrict__ netIn,
    unsigned short* __restrict__ netOut) {
  extern __shared__ char smem[];
  unsigned short* w0L = (unsigned short*)(smem + W0L_OFF);
  unsigned short* w1L = (unsigned short*)(smem + W1L_OFF);
  unsigned short* wsL = (unsigned short*)(smem + WSL_OFF);
  unsigned short* xnb = (unsigned short*)(smem + XN_OFF);
  unsigned short* hb  = (unsigned short*)(smem + HB_OFF);
  unsigned short* obb = (unsigned short*)(smem + OB_OFF);
  int* ibb            = (int*)(smem + MISC_OFF);
  const int w = (int)(threadIdx.x >> 5), lane = (int)(threadIdx.x & 31u);
  unsigned short* xn = xnb + w * MROWS * HD2;
  unsigned short* hl = hb  + w * MROWS * HDIM;
  unsigned short* ob = obb + w * MROWS * HDIM;
  int* il = ibb + w * MROWS;
  const int pt0 = blockIdx.x * PTS_PER_WG + w * MROWS;
  const int b = pt0 >> 16;

  stage_async(w0L, w0f, HD2 * HDIM);
  stage_async(w1L, w1f, HDIM * HDIM);
  stage_async(wsL, wsf, HD2 * HDIM);

  il[lane] = gidx[pt0 + lane];
  // x = concat(net, pooled-max) as bf16
  for (int t = lane; t < MROWS * HD2; t += 32) {
    int m = t >> 8, n = t & 255;
    if (n < HDIM) {
      xn[t] = netIn[(size_t)(pt0 + m) * HDIM + n];
    } else {
      int bin = il[m];
      xn[t] = f2bfu(decMax(bins[((size_t)b * NBINS + bin) * HDIM + (n - HDIM)]));
    }
  }
  wait_async0();
  __syncthreads();
  wave_resblock32(xn, hl, ob, w0L, w1L, wsL, b0, b1, netOut + (size_t)pt0 * HDIM);
}

// fc_c + scatter-mean accumulation (sums into d_out, counts into cnt); 16 pts/wave
__global__ __launch_bounds__(BLOCK) void k_head(
    const unsigned short* __restrict__ net,
    const int* __restrict__ gidx,
    const unsigned short* __restrict__ fccf, const float* __restrict__ fccb,
    float* __restrict__ outSum, float* __restrict__ cnt) {
  extern __shared__ char smem[];
  unsigned short* hbs = (unsigned short*)smem;                 // WAVES*16*128 bf16
  int* ib = (int*)(hbs + WAVES * 16 * HDIM);
  const int w = (int)(threadIdx.x >> 5), lane = (int)(threadIdx.x & 31u);
  unsigned short* hl = hbs + w * 16 * HDIM;
  int* il = ib + w * 16;
  const int pt0 = blockIdx.x * (WAVES * 16) + w * 16;
  const int b = pt0 >> 16;

  if (lane < 16) {
    int id = gidx[pt0 + lane];
    il[lane] = id;
    atomicAdd(cnt + (size_t)b * NBINS + id, 1.0f);
  }
  for (int t = lane; t < 16 * HDIM / 8; t += 32)      // async-stage net rows too
    async_ld16(hl + (size_t)t * 8, net + (size_t)pt0 * HDIM + (size_t)t * 8);
  wait_async0();
  __syncthreads();

  const int m16 = lane & 15;
  const bool hi = lane >= 16;
  const unsigned short* hrow = hl + m16 * HDIM;
  for (int nt = 0; nt < 8; ++nt) {
    fx8 acc = {};
#pragma unroll
    for (int kt = 0; kt < 4; ++kt) {
      bfx16 a = asA(loadA_pairs(hrow, kt, hi));
      bfx16 bb = loadB(fccf, kt, nt, 4, lane);
      acc = wmma_bf16(a, bb, acc);
    }
    int n = nt * 16 + m16;
    float bias = fccb[n];
#pragma unroll
    for (int r = 0; r < 8; ++r) {
      int m = (hi ? 8 : 0) + r;
      int bin = il[m];
      atomicAdd(outSum + ((size_t)b * HDIM + n) * NBINS + bin, acc[r] + bias);
    }
  }
}

// finalize scatter_mean: out /= max(cnt, 1)
__global__ void k_div(float* __restrict__ out, const float* __restrict__ cnt) {
  long total = (long)BATCH * HDIM * NBINS;
  for (long i = blockIdx.x * (long)blockDim.x + threadIdx.x; i < total;
       i += (long)gridDim.x * blockDim.x) {
    int bin = (int)(i & (NBINS - 1));
    int b = (int)(i >> 21);                      // 128*16384 = 2^21 per batch
    out[i] = out[i] / fmaxf(cnt[(size_t)b * NBINS + bin], 1.0f);
  }
}

// ---------------- launcher ----------------
extern "C" void kernel_launch(void* const* d_in, const int* in_sizes, int n_in,
                              void* d_out, int out_size, void* d_ws, size_t ws_size,
                              hipStream_t stream) {
  (void)in_sizes; (void)n_in; (void)out_size; (void)ws_size;
  const float* p    = (const float*)d_in[0];
  const float* fpw  = (const float*)d_in[1];
  const float* fpb  = (const float*)d_in[2];
  const float* w0   = (const float*)d_in[3];
  const float* b0   = (const float*)d_in[4];
  const float* w1   = (const float*)d_in[5];
  const float* b1   = (const float*)d_in[6];
  const float* wsm  = (const float*)d_in[7];
  const float* fcc  = (const float*)d_in[8];
  const float* fccb = (const float*)d_in[9];
  float* out = (float*)d_out;

  char* ws = (char*)d_ws;
  size_t off = 0;
  auto take = [&](size_t bytes) -> char* {
    char* r = ws + off;
    off = (off + bytes + 255) & ~(size_t)255;
    return r;
  };
  unsigned short* w0f  = (unsigned short*)take(5ull * HD2 * HDIM * 2);
  unsigned short* w1f  = (unsigned short*)take(5ull * HDIM * HDIM * 2);
  unsigned short* wsf  = (unsigned short*)take(5ull * HD2 * HDIM * 2);
  unsigned short* fccf = (unsigned short*)take((size_t)HDIM * HDIM * 2);
  int* gidx            = (int*)take((size_t)NPTS * 4);
  unsigned short* net  = (unsigned short*)take((size_t)NPTS * HDIM * 2);
  unsigned* bins       = (unsigned*)take((size_t)BATCH * NBINS * HDIM * 4);
  float* cnt           = (float*)take((size_t)BATCH * NBINS * 4);

  // repack weights to bf16 WMMA fragments (tiny; L2-resident afterwards)
  k_prep_frag<<<512, 256, 0, stream>>>(w0,  w0f,  HD2,  5);
  k_prep_frag<<<256, 256, 0, stream>>>(w1,  w1f,  HDIM, 5);
  k_prep_frag<<<512, 256, 0, stream>>>(wsm, wsf,  HD2,  5);
  k_prep_frag<<<128, 256, 0, stream>>>(fcc, fccf, HDIM, 1);

  const size_t sh_head = (size_t)WAVES * (16 * HDIM * 2 + 16 * 4);

  k_encode<<<NWG2, BLOCK, SH_BIG, stream>>>(p, fpw, fpb, w0f, b0, w1f, b1, wsf, gidx, net);

  for (int blk = 1; blk < 5; ++blk) {
    k_fill_u32<<<2048, 256, 0, stream>>>(bins, 0u, (long)BATCH * NBINS * HDIM);
    k_scatter_max<<<8192, 256, 0, stream>>>(net, gidx, bins);
    k_block<<<NWG2, BLOCK, SH_BIG, stream>>>(
        bins, gidx,
        w0f + (size_t)blk * HD2 * HDIM, b0 + blk * HDIM,
        w1f + (size_t)blk * HDIM * HDIM, b1 + blk * HDIM,
        wsf + (size_t)blk * HD2 * HDIM,
        net, net);
  }

  k_fill_u32<<<2048, 256, 0, stream>>>((unsigned*)out, 0u, (long)BATCH * HDIM * NBINS);
  k_fill_u32<<<64,   256, 0, stream>>>((unsigned*)cnt, 0u, (long)BATCH * NBINS);
  k_head<<<NWG_HEAD, BLOCK, sh_head, stream>>>(net, gidx, fccf, fccb, out, cnt);
  k_div<<<2048, 256, 0, stream>>>(out, cnt);
}
